// Struct2SeqGNN_22144851378807
// MI455X (gfx1250) — matrix operand order repaired and checked by reading, not compiled, not run on hardware
//
#include <hip/hip_runtime.h>
#include <hip/hip_bf16.h>
#include <math.h>

#define H 128
#define DG 16
#define NLAYER 4
#define NPN 50000
#define NLN 10000
#define EPP 500000
#define ELP 100000
#define EPL 100000
#define KDIM 272          // 2H + D
#define KPAD 288          // padded to 9*32
#define KC 9              // K chunks of 32
#define NC 8              // N chunks of 16
#define MT 4              // M-tiles (of 16 edges) per wave -> 64 edges/block
#define MAT_ELEMS (KC*NC*32*16)   // 36864 bf16 per packed matrix

typedef __attribute__((ext_vector_type(16))) __bf16 v16bf;
typedef __attribute__((ext_vector_type(8)))  __bf16 v8bf;
typedef __attribute__((ext_vector_type(8)))  float  v8f;

// ---------------------------------------------------------------- utilities
__global__ void zero_f(float* p, long n) {
    long i = (long)blockIdx.x * blockDim.x + threadIdx.x;
    long stride = (long)gridDim.x * blockDim.x;
    for (; i < n; i += stride) p[i] = 0.0f;
}

// out[n,t] = sum_k x[n,k] * W[k,t] + b[t]   (k<6, t<128)
__global__ void embed_kernel(const float* __restrict__ x, const float* __restrict__ W,
                             const float* __restrict__ b, float* __restrict__ out, int N) {
    int n = blockIdx.x;
    int t = threadIdx.x;
    if (n >= N) return;
    float acc = b[t];
#pragma unroll
    for (int k = 0; k < 6; ++k)
        acc += x[(size_t)n * 6 + k] * W[k * H + t];
    out[(size_t)n * H + t] = acc;
}

// GaussianSmearing -> bf16, [E,16]
__global__ void rbf_kernel(const float* __restrict__ d, __bf16* __restrict__ out, int E) {
    int tid = blockIdx.x * blockDim.x + threadIdx.x;
    if (tid >= E * DG) return;
    int e = tid >> 4, g = tid & 15;
    const float step  = 8.0f / 15.0f;
    const float coeff = -0.5f / (step * step);
    float x = d[e] - step * (float)g;
    out[tid] = (__bf16)__expf(coeff * x * x);
}

// Pack Wf/Ws (L,3,272,128) f32 -> bf16 WMMA-B-operand tile order, K padded to 288.
// dest element index: mat*MAT_ELEMS + ((kc*NC+nc)*32 + L)*16 + j
// value = W[l,r, k = kc*32 + 16*(L/16) + j, n = nc*16 + L%16]
__global__ void pack_weights(const float* __restrict__ Wf, const float* __restrict__ Ws,
                             __bf16* __restrict__ packed) {
    int tid = blockIdx.x * blockDim.x + threadIdx.x;
    const int total = NLAYER * 3 * 2 * MAT_ELEMS;
    if (tid >= total) return;
    int mat = tid / MAT_ELEMS;
    int rem = tid % MAT_ELEMS;
    int j  = rem & 15;
    int L  = (rem >> 4) & 31;
    int t  = rem >> 9;          // kc*NC + nc
    int nc = t & 7, kc = t >> 3;
    int k = kc * 32 + ((L >> 4) << 4) + j;
    int n = nc * 16 + (L & 15);
    int lr = mat >> 1;          // l*3 + r
    const float* src = (mat & 1) ? Ws : Wf;
    float v = (k < KDIM) ? src[((size_t)lr * KDIM + k) * H + n] : 0.0f;
    packed[tid] = (__bf16)v;
}

// ---------------------------------------------------------------- edge GEMM (WMMA)
// One wave (32 threads) processes 64 edges (MT=4 tiles of 16).
// z = [x_dst[dst] (128) | x_src[src] (128) | rbf (16) | pad(16)] in bf16, LDS staged.
// m = sigmoid(z@Wf + bf) * softplus(z@Ws + bs); atomicAdd into agg[dst].
// B fragments loaded once per (nc,kc) and reused across 4 M-tiles (4x less
// weight traffic; 8 independent WMMA chains per step).
__global__ __launch_bounds__(32)
void edge_msg_kernel(const float* __restrict__ xdst, const float* __restrict__ xsrc,
                     const int* __restrict__ ei, const __bf16* __restrict__ rbf,
                     const __bf16* __restrict__ packWf, const __bf16* __restrict__ packWs,
                     const float* __restrict__ bfv, const float* __restrict__ bsv,
                     float* __restrict__ agg, int E) {
    __shared__ __align__(16) __bf16 zrow[16 * MT * KPAD];   // 36864 B
    __shared__ int ssrc[16 * MT];
    __shared__ int sdst[16 * MT];

    const int lane = threadIdx.x;
    const int e0 = blockIdx.x * (16 * MT);
    if (e0 >= E) return;

    // ---- load edge indices (clamped for tail blocks)
#pragma unroll
    for (int t = 0; t < MT; ++t) {
        if (lane < 16) {
            int e = e0 + t * 16 + lane;
            int ec = e < E ? e : (E - 1);
            ssrc[t * 16 + lane] = ei[ec];        // ei[0] = src
            sdst[t * 16 + lane] = ei[E + ec];    // ei[1] = dst
        }
    }
    __syncthreads();

    // ---- stage 64 z-rows into LDS as bf16
    for (int m = 0; m < 16 * MT; ++m) {
        int dn = sdst[m];
        int sn = ssrc[m];
        float4 fd = *(const float4*)(xdst + (size_t)dn * H + lane * 4);
        float4 fs = *(const float4*)(xsrc + (size_t)sn * H + lane * 4);
        __bf16* zr = &zrow[m * KPAD];
        zr[lane * 4 + 0] = (__bf16)fd.x;
        zr[lane * 4 + 1] = (__bf16)fd.y;
        zr[lane * 4 + 2] = (__bf16)fd.z;
        zr[lane * 4 + 3] = (__bf16)fd.w;
        zr[H + lane * 4 + 0] = (__bf16)fs.x;
        zr[H + lane * 4 + 1] = (__bf16)fs.y;
        zr[H + lane * 4 + 2] = (__bf16)fs.z;
        zr[H + lane * 4 + 3] = (__bf16)fs.w;
        int e = e0 + m;
        int ec = e < E ? e : (E - 1);
        // rbf (lanes 0..15) and zero pad K=272..287 (lanes 16..31)
        zr[256 + lane] = (lane < 16) ? rbf[(size_t)ec * DG + lane] : (__bf16)0.0f;
    }
    __syncthreads();

    const int mrow  = lane & 15;
    const int hi    = lane >> 4;         // 0 or 1
    const int abase = hi << 4;           // A operand K sub-offset (+16 for upper half-wave)

#pragma unroll 1
    for (int nc = 0; nc < NC; ++nc) {
        v8f accF[MT];
        v8f accS[MT];
#pragma unroll
        for (int t = 0; t < MT; ++t) { accF[t] = (v8f){}; accS[t] = (v8f){}; }

#pragma unroll
        for (int kc = 0; kc < KC; ++kc) {
            // ---- B fragments from packed global (coalesced 32B/lane), shared by MT tiles
            size_t boff = ((size_t)(kc * NC + nc) * 32 + lane) * 16;
            v16bf Bf = *(const v16bf*)(packWf + boff);
            v16bf Bs = *(const v16bf*)(packWs + boff);
#pragma unroll
            for (int t = 0; t < MT; ++t) {
                // ---- A fragment from LDS (16-bit 16x32 A layout)
                const __bf16* zp = &zrow[(t * 16 + mrow) * KPAD + kc * 32 + abase];
                v8bf a0 = *(const v8bf*)zp;          // K = kc*32 + abase + 0..7
                v8bf a1 = *(const v8bf*)(zp + 8);    // K = kc*32 + abase + 8..15
                v16bf A;
#pragma unroll
                for (int j = 0; j < 8; ++j) { A[j] = a0[j]; A[j + 8] = a1[j]; }
                accF[t] = __builtin_amdgcn_wmma_f32_16x16x32_bf16(false, A, false, Bf,
                                                                  (short)0, accF[t], false, false);
                accS[t] = __builtin_amdgcn_wmma_f32_16x16x32_bf16(false, A, false, Bs,
                                                                  (short)0, accS[t], false, false);
            }
        }
        // ---- epilogue: bias + sigmoid*softplus + scatter-add
        int n = nc * 16 + (lane & 15);
        float bF = bfv[n];
        float bS = bsv[n];
#pragma unroll
        for (int t = 0; t < MT; ++t) {
#pragma unroll
            for (int r = 0; r < 8; ++r) {
                int m = r + (hi << 3);
                int e = e0 + t * 16 + m;
                float f = accF[t][r] + bF;
                float s = accS[t][r] + bS;
                float sig = 1.0f / (1.0f + __expf(-f));
                float sp  = (s > 20.0f) ? s : log1pf(__expf(s));
                float msg = sig * sp;
                if (e < E) {
                    int dn = sdst[t * 16 + m];
                    atomicAdd(&agg[(size_t)dn * H + n], msg);
                }
            }
        }
    }
}

// ---------------------------------------------------------------- node updates
__device__ __forceinline__ float block_reduce_sum(float v, float* red) {
    int t = threadIdx.x;
    red[t] = v; __syncthreads();
    for (int s = 64; s > 0; s >>= 1) {
        if (t < s) red[t] += red[t + s];
        __syncthreads();
    }
    float r = red[0]; __syncthreads();
    return r;
}

__device__ __forceinline__ float cg_post(float aggv, float x, float bnw, float bnb,
                                         float lnw, float lnb, float* red) {
    const float bnscale = rsqrtf(1.0f + 1e-5f);
    float v = aggv * (bnw * bnscale) + bnb + x;     // BN(eval) + CGConv residual
    float mu  = block_reduce_sum(v, red) * (1.0f / (float)H);
    float d   = v - mu;
    float var = block_reduce_sum(d * d, red) * (1.0f / (float)H);
    float ln  = d * rsqrtf(var + 1e-5f) * lnw + lnb;
    float rl  = ln > 0.0f ? ln : 0.0f;
    return rl + x;                                   // block residual
}

__global__ __launch_bounds__(H)
void node_update_p(const float* __restrict__ xin,
                   const float* __restrict__ agg0, const float* __restrict__ agg1,
                   const float* __restrict__ bnw0, const float* __restrict__ bnb0,
                   const float* __restrict__ lnw0, const float* __restrict__ lnb0,
                   const float* __restrict__ bnw1, const float* __restrict__ bnb1,
                   const float* __restrict__ lnw1, const float* __restrict__ lnb1,
                   float* __restrict__ xout, int N) {
    __shared__ float red[H];
    int n = blockIdx.x, t = threadIdx.x;
    if (n >= N) return;
    size_t idx = (size_t)n * H + t;
    float x = xin[idx];
    float p0 = cg_post(agg0[idx], x, bnw0[t], bnb0[t], lnw0[t], lnb0[t], red);
    float p1 = cg_post(agg1[idx], x, bnw1[t], bnb1[t], lnw1[t], lnb1[t], red);
    xout[idx] = p0 + p1;
}

__global__ __launch_bounds__(H)
void node_update_l(const float* __restrict__ xin, const float* __restrict__ agg0,
                   const float* __restrict__ bnw0, const float* __restrict__ bnb0,
                   const float* __restrict__ lnw0, const float* __restrict__ lnb0,
                   float* __restrict__ xout, int N) {
    __shared__ float red[H];
    int n = blockIdx.x, t = threadIdx.x;
    if (n >= N) return;
    size_t idx = (size_t)n * H + t;
    float x = xin[idx];
    xout[idx] = cg_post(agg0[idx], x, bnw0[t], bnb0[t], lnw0[t], lnb0[t], red);
}

// ---------------------------------------------------------------- head
__global__ __launch_bounds__(H)
void head_kernel(const float* __restrict__ xin,
                 const float* __restrict__ lnow, const float* __restrict__ lnob,
                 const float* __restrict__ fcw, const float* __restrict__ fcb,
                 float* __restrict__ out, int N) {
    __shared__ float red[H];
    __shared__ float yl[H];
    int n = blockIdx.x, t = threadIdx.x;
    if (n >= N) return;
    float x = xin[(size_t)n * H + t];
    float mu  = block_reduce_sum(x, red) * (1.0f / (float)H);
    float d   = x - mu;
    float var = block_reduce_sum(d * d, red) * (1.0f / (float)H);
    yl[t] = d * rsqrtf(var + 1e-5f) * lnow[t] + lnob[t];
    __syncthreads();
    if (t < 21) {
        float acc = fcb[t];
#pragma unroll 8
        for (int k = 0; k < H; ++k) acc += yl[k] * fcw[k * 21 + t];
        out[(size_t)n * 21 + t] = acc;
    }
}

// ---------------------------------------------------------------- host driver
extern "C" void kernel_launch(void* const* d_in, const int* in_sizes, int n_in,
                              void* d_out, int out_size, void* d_ws, size_t ws_size,
                              hipStream_t stream) {
    (void)in_sizes; (void)n_in; (void)out_size; (void)ws_size;
    const float* x_protein = (const float*)d_in[0];
    const float* x_ligand  = (const float*)d_in[1];
    const int*   ei_pp     = (const int*)d_in[2];
    const float* ea_pp     = (const float*)d_in[3];
    const int*   ei_lp     = (const int*)d_in[4];
    const float* ea_lp     = (const float*)d_in[5];
    const int*   ei_pl     = (const int*)d_in[6];
    const float* ea_pl     = (const float*)d_in[7];
    const float* Wp = (const float*)d_in[8];
    const float* bp = (const float*)d_in[9];
    const float* Wl = (const float*)d_in[10];
    const float* bl = (const float*)d_in[11];
    const float* Wf = (const float*)d_in[12];
    const float* bfv = (const float*)d_in[13];
    const float* Ws = (const float*)d_in[14];
    const float* bsv = (const float*)d_in[15];
    const float* bn_w = (const float*)d_in[16];
    const float* bn_b = (const float*)d_in[17];
    const float* ln_w = (const float*)d_in[18];
    const float* ln_b = (const float*)d_in[19];
    const float* lno_w = (const float*)d_in[20];
    const float* lno_b = (const float*)d_in[21];
    const float* fc_w = (const float*)d_in[22];
    const float* fc_b = (const float*)d_in[23];

    char* ws = (char*)d_ws;
    size_t off = 0;
    auto salloc = [&](size_t bytes) { size_t o = off; off = (off + bytes + 255) & ~(size_t)255; return o; };
    float* xpA   = (float*)(ws + salloc((size_t)NPN * H * 4));
    float* xpB   = (float*)(ws + salloc((size_t)NPN * H * 4));
    float* xlA   = (float*)(ws + salloc((size_t)NLN * H * 4));
    float* xlB   = (float*)(ws + salloc((size_t)NLN * H * 4));
    float* aggPP = (float*)(ws + salloc((size_t)NPN * H * 4));
    float* aggLP = (float*)(ws + salloc((size_t)NPN * H * 4));
    float* aggPL = (float*)(ws + salloc((size_t)NLN * H * 4));
    __bf16* packW = (__bf16*)(ws + salloc((size_t)NLAYER * 3 * 2 * MAT_ELEMS * 2));
    __bf16* rbfPP = (__bf16*)(ws + salloc((size_t)EPP * DG * 2));
    __bf16* rbfLP = (__bf16*)(ws + salloc((size_t)ELP * DG * 2));
    __bf16* rbfPL = (__bf16*)(ws + salloc((size_t)EPL * DG * 2));

    // embeddings
    embed_kernel<<<NPN, H, 0, stream>>>(x_protein, Wp, bp, xpA, NPN);
    embed_kernel<<<NLN, H, 0, stream>>>(x_ligand, Wl, bl, xlA, NLN);
    // RBF expansion (reused across layers)
    rbf_kernel<<<(EPP * DG + 255) / 256, 256, 0, stream>>>(ea_pp, rbfPP, EPP);
    rbf_kernel<<<(ELP * DG + 255) / 256, 256, 0, stream>>>(ea_lp, rbfLP, ELP);
    rbf_kernel<<<(EPL * DG + 255) / 256, 256, 0, stream>>>(ea_pl, rbfPL, EPL);
    // pack weights to WMMA bf16 layout
    {
        int total = NLAYER * 3 * 2 * MAT_ELEMS;
        pack_weights<<<(total + 255) / 256, 256, 0, stream>>>(Wf, Ws, packW);
    }

    const int EB = 16 * MT;   // edges per block
    float* xp_cur = xpA; float* xp_nxt = xpB;
    float* xl_cur = xlA; float* xl_nxt = xlB;
    for (int l = 0; l < NLAYER; ++l) {
        zero_f<<<4096, 256, 0, stream>>>(aggPP, (long)NPN * H);
        zero_f<<<4096, 256, 0, stream>>>(aggLP, (long)NPN * H);
        zero_f<<<4096, 256, 0, stream>>>(aggPL, (long)NLN * H);
        int lr0 = l * 3 + 0, lr1 = l * 3 + 1, lr2 = l * 3 + 2;
        // protein <- protein
        edge_msg_kernel<<<(EPP + EB - 1) / EB, 32, 0, stream>>>(
            xp_cur, xp_cur, ei_pp, rbfPP,
            packW + (size_t)(lr0 * 2 + 0) * MAT_ELEMS, packW + (size_t)(lr0 * 2 + 1) * MAT_ELEMS,
            bfv + (size_t)lr0 * H, bsv + (size_t)lr0 * H, aggPP, EPP);
        // protein <- ligand
        edge_msg_kernel<<<(ELP + EB - 1) / EB, 32, 0, stream>>>(
            xp_cur, xl_cur, ei_lp, rbfLP,
            packW + (size_t)(lr1 * 2 + 0) * MAT_ELEMS, packW + (size_t)(lr1 * 2 + 1) * MAT_ELEMS,
            bfv + (size_t)lr1 * H, bsv + (size_t)lr1 * H, aggLP, ELP);
        // ligand <- protein
        edge_msg_kernel<<<(EPL + EB - 1) / EB, 32, 0, stream>>>(
            xl_cur, xp_cur, ei_pl, rbfPL,
            packW + (size_t)(lr2 * 2 + 0) * MAT_ELEMS, packW + (size_t)(lr2 * 2 + 1) * MAT_ELEMS,
            bfv + (size_t)lr2 * H, bsv + (size_t)lr2 * H, aggPL, EPL);
        // node updates
        node_update_p<<<NPN, H, 0, stream>>>(
            xp_cur, aggPP, aggLP,
            bn_w + (size_t)lr0 * H, bn_b + (size_t)lr0 * H, ln_w + (size_t)lr0 * H, ln_b + (size_t)lr0 * H,
            bn_w + (size_t)lr1 * H, bn_b + (size_t)lr1 * H, ln_w + (size_t)lr1 * H, ln_b + (size_t)lr1 * H,
            xp_nxt, NPN);
        node_update_l<<<NLN, H, 0, stream>>>(
            xl_cur, aggPL,
            bn_w + (size_t)lr2 * H, bn_b + (size_t)lr2 * H, ln_w + (size_t)lr2 * H, ln_b + (size_t)lr2 * H,
            xl_nxt, NLN);
        float* tp = xp_cur; xp_cur = xp_nxt; xp_nxt = tp;
        float* tl = xl_cur; xl_cur = xl_nxt; xl_nxt = tl;
    }
    head_kernel<<<NPN, H, 0, stream>>>(xp_cur, lno_w, lno_b, fc_w, fc_b, (float*)d_out, NPN);
}
